// MHSABlock_4294967296212
// MI455X (gfx1250) — matrix operand order repaired and checked by reading, not compiled, not run on hardware
//
#include <hip/hip_runtime.h>

// ---------------------------------------------------------------------------
// MHSA block for gfx1250 (MI455X).  All matmuls via v_wmma_f32_16x16x32_f16.
// b=2, c=256, heads=8, dk=32, N=4096 tokens (64x64).
// ---------------------------------------------------------------------------

typedef __attribute__((ext_vector_type(16))) _Float16 v16h;
typedef __attribute__((ext_vector_type(8)))  _Float16 v8h;
typedef __attribute__((ext_vector_type(8)))  float    v8f;
typedef __attribute__((ext_vector_type(4)))  float    v4f;

#define BATCH 2
#define CDIM  256
#define HEADS 8
#define DK    32
#define NTOK  4096           // tokens per batch (64*64)
#define TTOT  8192           // tokens total (both batches)

// ---- WMMA operand loaders (16x16x32 f16, this code's contiguous k order for A and B) --------------------
// A (16x32, f16): lane&15 = row M; lanes<16 hold K 0..15,
//                 lanes>=16 hold K 16..31; 2 halves per VGPR.
__device__ __forceinline__ v16h loadA16(const _Float16* p, int stride) {
  const int lane = threadIdx.x & 31;
  const _Float16* q = p + (long)(lane & 15) * stride + ((lane >> 4) << 4);
  v8h lo = *(const v8h*)q;          // K +0..7   (or +16..23)
  v8h hi = *(const v8h*)(q + 8);    // K +8..15  (or +24..31)
  v16h a;
#pragma unroll
  for (int i = 0; i < 8; ++i) { a[i] = lo[i]; a[i + 8] = hi[i]; }
  return a;
}

// B (32x16, f16): lane&15 = column N; lanes<16 hold K 0..15,
//                 lanes>=16 hold K 16..31; contiguous 32 bytes per lane.
__device__ __forceinline__ v16h loadB16(const _Float16* p, int stride) {
  const int lane = threadIdx.x & 31;
  return *(const v16h*)(p + (long)(lane & 15) * stride + ((lane >> 4) << 4));
}

__device__ __forceinline__ v8f wmma16(v16h a, v16h b, v8f c) {
  return __builtin_amdgcn_wmma_f32_16x16x32_f16(false, a, false, b,
                                                (short)0, c, false, false);
}

// ---- 2D positional encoding (wzlxjtu variant), computed on the fly --------
__device__ __forceinline__ float pe2d(int c, int n) {
  const int hh = n >> 6, ww = n & 63;
  int i, pos;
  if (c < 128) { i = c >> 1;         pos = ww; }
  else         { i = (c - 128) >> 1; pos = hh; }
  // div_term[i] = exp(-2*i*log(10000)/128)
  const float arg = (float)pos * __expf(-0.14391156816013644f * (float)i);
  return (c & 1) ? __cosf(arg) : __sinf(arg);
}

// ---- kernel 1: weights f32 -> f16 (row-major [c_out][c_in] preserved) -----
__global__ __launch_bounds__(256) void wcvt_kernel(const float* a, const float* b,
                                                   const float* c, const float* d,
                                                   _Float16* o) {
  const int i = blockIdx.x * 256 + threadIdx.x;      // 0..65535
  o[i]          = (_Float16)a[i];
  o[65536 + i]  = (_Float16)b[i];
  o[131072 + i] = (_Float16)c[i];
  o[196608 + i] = (_Float16)d[i];
}

// ---- kernel 2: tok[t][c] = f16(x[b][c][n] + pe(c,n)) via LDS transpose ----
__global__ __launch_bounds__(256) void tok_kernel(const float* __restrict__ x,
                                                  _Float16* __restrict__ tok) {
  __shared__ _Float16 tile[64][72];                  // padded vs bank conflicts
  const int b = blockIdx.z, c0 = blockIdx.y * 64, n0 = blockIdx.x * 64;
  const int tid = threadIdx.x;
#pragma unroll 4
  for (int it = 0; it < 16; ++it) {
    const int idx = it * 256 + tid;
    const int cc = idx >> 6, nn = idx & 63;
    const int c = c0 + cc, n = n0 + nn;
    const float xv = x[((long)(b * CDIM + c)) * NTOK + n];   // coalesced in n
    tile[cc][nn] = (_Float16)(xv + pe2d(c, n));
  }
  __syncthreads();
#pragma unroll 4
  for (int it = 0; it < 16; ++it) {
    const int idx = it * 256 + tid;
    const int cc = idx & 63, nn = idx >> 6;
    tok[((long)(b * NTOK + n0 + nn)) * CDIM + c0 + cc] = tile[cc][nn]; // coalesced in c
  }
}

// ---- kernel 3: Q,K projection.  C-tile M = out-channel so that the store
//      into q[token][256] / k[token][256] is per-lane contiguous (v8h b128).
//      Qt = Wq(A: ch x k) @ tok^T(B: k x tokens).  1/sqrt(dk) folded into Q.
__global__ __launch_bounds__(256) void qk_proj_kernel(const _Float16* __restrict__ tok,
                                                      const _Float16* __restrict__ wq16,
                                                      const _Float16* __restrict__ wk16,
                                                      const float* __restrict__ bq,
                                                      const float* __restrict__ bk,
                                                      _Float16* __restrict__ qm,
                                                      _Float16* __restrict__ km) {
  const bool isQ = (blockIdx.y == 0);
  const _Float16* W = isQ ? wq16 : wk16;
  const float* bias = isQ ? bq : bk;
  _Float16* dst     = isQ ? qm : km;
  const float scale = isQ ? 0.17677669529663687f : 1.0f;   // 1/sqrt(32)

  const int wid  = blockIdx.x * 8 + (threadIdx.x >> 5);    // 2048 waves
  const int lane = threadIdx.x & 31;
  const int  mb = (wid >> 8) * 32;                         // channel base (0..224)
  const long tb = (long)(wid & 255) * 32;                  // token base  (0..8160)

  v8f zero = {};
  v8f acc[2][2];
#pragma unroll
  for (int i = 0; i < 2; ++i)
#pragma unroll
    for (int j = 0; j < 2; ++j) acc[i][j] = zero;

#pragma unroll 2
  for (int k0 = 0; k0 < CDIM; k0 += 32) {
    v16h a0 = loadA16(W + (long)mb * CDIM + k0, CDIM);
    v16h a1 = loadA16(W + (long)(mb + 16) * CDIM + k0, CDIM);
    v16h b0 = loadB16(tok + tb * CDIM + k0, CDIM);
    v16h b1 = loadB16(tok + (tb + 16) * CDIM + k0, CDIM);
    acc[0][0] = wmma16(a0, b0, acc[0][0]);
    acc[0][1] = wmma16(a0, b1, acc[0][1]);
    acc[1][0] = wmma16(a1, b0, acc[1][0]);
    acc[1][1] = wmma16(a1, b1, acc[1][1]);
  }
#pragma unroll
  for (int i = 0; i < 2; ++i) {
    const int chb = mb + 16 * i + ((lane >> 4) << 3);
    const float* bp = bias + chb;
#pragma unroll
    for (int j = 0; j < 2; ++j) {
      const long t = tb + 16 * j + (lane & 15);
      v8h ph;
#pragma unroll
      for (int r = 0; r < 8; ++r)
        ph[r] = (_Float16)((acc[i][j][r] + bp[r]) * scale);
      *(v8h*)(dst + t * CDIM + chb) = ph;
    }
  }
}

// ---- kernel 4: V projection, stored transposed vt[b][ch][token].
//      V = tok(A: tokens x k) @ Wv^T(B: k x ch); C-tile M = token ->
//      contiguous store along the token axis of vt.
__global__ __launch_bounds__(256) void v_proj_kernel(const _Float16* __restrict__ tok,
                                                     const _Float16* __restrict__ wv16,
                                                     const float* __restrict__ bv,
                                                     _Float16* __restrict__ vt) {
  const int wid  = blockIdx.x * 8 + (threadIdx.x >> 5);    // 2048 waves
  const int lane = threadIdx.x & 31;
  const long tb = (long)(wid >> 3) * 32;                   // token base
  const int chb0 = (wid & 7) * 32;                         // channel base

  v8f zero = {};
  v8f acc[2][2];
#pragma unroll
  for (int i = 0; i < 2; ++i)
#pragma unroll
    for (int j = 0; j < 2; ++j) acc[i][j] = zero;

#pragma unroll 2
  for (int k0 = 0; k0 < CDIM; k0 += 32) {
    v16h a0 = loadA16(tok + tb * CDIM + k0, CDIM);
    v16h a1 = loadA16(tok + (tb + 16) * CDIM + k0, CDIM);
    v16h b0 = loadB16(wv16 + (long)chb0 * CDIM + k0, CDIM);
    v16h b1 = loadB16(wv16 + (long)(chb0 + 16) * CDIM + k0, CDIM);
    acc[0][0] = wmma16(a0, b0, acc[0][0]);
    acc[0][1] = wmma16(a0, b1, acc[0][1]);
    acc[1][0] = wmma16(a1, b0, acc[1][0]);
    acc[1][1] = wmma16(a1, b1, acc[1][1]);
  }
#pragma unroll
  for (int i = 0; i < 2; ++i) {
    const long t0 = tb + 16 * i + ((lane >> 4) << 3);
    const long vb = (t0 >> 12) * CDIM;                     // batch*256
    const long n0 = t0 & (NTOK - 1);
#pragma unroll
    for (int j = 0; j < 2; ++j) {
      const int ch = chb0 + 16 * j + (lane & 15);
      const float bvv = bv[ch];
      v8h ph;
#pragma unroll
      for (int r = 0; r < 8; ++r)
        ph[r] = (_Float16)(acc[i][j][r] + bvv);
      *(v8h*)(vt + (vb + ch) * (long)NTOK + n0) = ph;
    }
  }
}

// ---- kernel 5: flash attention.  One wave = (batch, head, 16 queries).
//      S^T = K(A) @ Q^T(B)  -> softmax over keys is within-lane.
//      O^T = V^T(A) @ P^T(B) -> concat store is per-lane contiguous.
__global__ __launch_bounds__(256) void attn_kernel(const _Float16* __restrict__ qmat,
                                                   const _Float16* __restrict__ kmat,
                                                   const _Float16* __restrict__ vt,
                                                   _Float16* __restrict__ concat) {
  const int wid  = blockIdx.x * 8 + (threadIdx.x >> 5);    // 4096 waves
  const int lane = threadIdx.x & 31;
  const int b  = wid >> 11;
  const int rm = wid & 2047;
  const int h  = rm >> 8;
  const int qi = rm & 255;
  const long gq = (long)b * NTOK + qi * 16;
  const int co = h * DK;

  const v16h Bq = loadB16(qmat + gq * CDIM + co, CDIM);    // Q^T, loaded once

  float m = -1e30f, lsum = 0.0f;
  v8f O0 = {}, O1 = {};

  for (int kb = 0; kb < NTOK; kb += 32) {
    const long gk = (long)b * NTOK + kb;
    v16h Ak0 = loadA16(kmat + gk * CDIM + co, CDIM);
    v16h Ak1 = loadA16(kmat + (gk + 16) * CDIM + co, CDIM);
    v8f z = {};
    v8f S0 = wmma16(Ak0, Bq, z);     // keys kb+0..15  x queries
    v8f S1 = wmma16(Ak1, Bq, z);     // keys kb+16..31 x queries

    // ---- online softmax over keys (per query column; lanes l,l+16 synced)
    float bm = -1e30f;
#pragma unroll
    for (int r = 0; r < 8; ++r) bm = fmaxf(bm, fmaxf(S0[r], S1[r]));
    bm = fmaxf(bm, __shfl_xor(bm, 16, 32));
    const float mnew = fmaxf(m, bm);
    const float corr = __expf(m - mnew);

    float p0[8], p1[8];
    float ps = 0.0f;
#pragma unroll
    for (int r = 0; r < 8; ++r) {
      p0[r] = __expf(S0[r] - mnew);
      p1[r] = __expf(S1[r] - mnew);
      ps += p0[r] + p1[r];
    }
    ps += __shfl_xor(ps, 16, 32);
    lsum = lsum * corr + ps;
#pragma unroll
    for (int r = 0; r < 8; ++r) { O0[r] *= corr; O1[r] *= corr; }
    m = mnew;

    // ---- build P^T as B-operand: lane<16 needs keys 0..15, lane>=16 keys 16..31
    float o0[8], o1[8];
#pragma unroll
    for (int r = 0; r < 8; ++r) {
      o0[r] = __shfl_xor(p0[r], 16, 32);
      o1[r] = __shfl_xor(p1[r], 16, 32);
    }
    const bool hi = (lane & 16) != 0;
    v16h Pt;
#pragma unroll
    for (int r = 0; r < 8; ++r) {
      Pt[r]     = (_Float16)(hi ? o1[r] : p0[r]);
      Pt[r + 8] = (_Float16)(hi ? p1[r] : o0[r]);
    }

    // ---- O^T += V^T @ P^T  (two dk tiles)
    const _Float16* vbase = vt + ((long)b * CDIM + co) * NTOK + kb;
    v16h Av0 = loadA16(vbase, NTOK);
    v16h Av1 = loadA16(vbase + 16L * NTOK, NTOK);
    O0 = wmma16(Av0, Pt, O0);
    O1 = wmma16(Av1, Pt, O1);
  }

  const float inv = 1.0f / lsum;
  _Float16* cp = concat + (gq + (lane & 15)) * CDIM + co + ((lane >> 4) << 3);
  v8h ph0, ph1;
#pragma unroll
  for (int r = 0; r < 8; ++r) {
    ph0[r] = (_Float16)(O0[r] * inv);
    ph1[r] = (_Float16)(O1[r] * inv);
  }
  *(v8h*)cp = ph0;          // channels co+0..15 slice for this lane
  *(v8h*)(cp + 16) = ph1;   // channels co+16..31
}

// ---- kernel 6: output projection, stores final [b][c][h][w] f32.
//      proj = concat(A: tokens x k) @ Wo^T(B) + bo; C-tile M = token ->
//      contiguous f32 stores along the n (=h*w) axis.
__global__ __launch_bounds__(256) void out_proj_kernel(const _Float16* __restrict__ concat,
                                                       const _Float16* __restrict__ wo16,
                                                       const float* __restrict__ bo,
                                                       float* __restrict__ out) {
  const int wid  = blockIdx.x * 8 + (threadIdx.x >> 5);    // 2048 waves
  const int lane = threadIdx.x & 31;
  const long tb = (long)(wid >> 3) * 32;
  const int chb0 = (wid & 7) * 32;

  v8f zero = {};
  v8f acc[2][2];
#pragma unroll
  for (int i = 0; i < 2; ++i)
#pragma unroll
    for (int j = 0; j < 2; ++j) acc[i][j] = zero;

#pragma unroll 2
  for (int k0 = 0; k0 < CDIM; k0 += 32) {
    v16h a0 = loadA16(concat + tb * CDIM + k0, CDIM);
    v16h a1 = loadA16(concat + (tb + 16) * CDIM + k0, CDIM);
    v16h b0 = loadB16(wo16 + (long)chb0 * CDIM + k0, CDIM);
    v16h b1 = loadB16(wo16 + (long)(chb0 + 16) * CDIM + k0, CDIM);
    acc[0][0] = wmma16(a0, b0, acc[0][0]);
    acc[0][1] = wmma16(a0, b1, acc[0][1]);
    acc[1][0] = wmma16(a1, b0, acc[1][0]);
    acc[1][1] = wmma16(a1, b1, acc[1][1]);
  }
#pragma unroll
  for (int i = 0; i < 2; ++i) {
    const long t0 = tb + 16 * i + ((lane >> 4) << 3);
    const long bb = t0 >> 12;
    const long n0 = t0 & (NTOK - 1);
#pragma unroll
    for (int j = 0; j < 2; ++j) {
      const int ch = chb0 + 16 * j + (lane & 15);
      const float bov = bo[ch];
      float* op = out + ((bb * CDIM + ch) * NTOK) + n0;
      v4f lo4, hi4;
#pragma unroll
      for (int r = 0; r < 4; ++r) {
        lo4[r] = acc[i][j][r] + bov;
        hi4[r] = acc[i][j][r + 4] + bov;
      }
      *(v4f*)op = lo4;
      *(v4f*)(op + 4) = hi4;
    }
  }
}

// ---------------------------------------------------------------------------
extern "C" void kernel_launch(void* const* d_in, const int* in_sizes, int n_in,
                              void* d_out, int out_size, void* d_ws, size_t ws_size,
                              hipStream_t stream) {
  const float* x  = (const float*)d_in[0];
  const float* Wq = (const float*)d_in[1];
  const float* bq = (const float*)d_in[2];
  const float* Wk = (const float*)d_in[3];
  const float* bk = (const float*)d_in[4];
  const float* Wv = (const float*)d_in[5];
  const float* bv = (const float*)d_in[6];
  const float* Wo = (const float*)d_in[7];
  const float* bo = (const float*)d_in[8];
  float* out = (float*)d_out;

  // workspace carving (all f16, all >=32B aligned)
  _Float16* wsh  = (_Float16*)d_ws;
  _Float16* wq16 = wsh;                      //  65536
  _Float16* wk16 = wq16 + 65536;
  _Float16* wv16 = wk16 + 65536;
  _Float16* wo16 = wv16 + 65536;
  _Float16* tok  = wo16 + 65536;             // [8192][256]
  _Float16* qm   = tok + (long)TTOT * CDIM;  // [8192][256]
  _Float16* km   = qm  + (long)TTOT * CDIM;  // [8192][256]
  _Float16* vtm  = km  + (long)TTOT * CDIM;  // [2][256][4096]
  _Float16* cm   = vtm + (long)TTOT * CDIM;  // [8192][256]

  wcvt_kernel<<<256, 256, 0, stream>>>(Wq, Wk, Wv, Wo, wq16);
  tok_kernel<<<dim3(64, 4, 2), 256, 0, stream>>>(x, tok);
  qk_proj_kernel<<<dim3(256, 2), 256, 0, stream>>>(tok, wq16, wk16, bq, bk, qm, km);
  v_proj_kernel<<<256, 256, 0, stream>>>(tok, wv16, bv, vtm);
  attn_kernel<<<512, 256, 0, stream>>>(qm, km, vtm, cm);
  out_proj_kernel<<<256, 256, 0, stream>>>(cm, wo16, bo, out);
}
